// BaseHSMM_20504173871544
// MI455X (gfx1250) — matrix operand order repaired and checked by reading, not compiled, not run on hardware
//
#include <hip/hip_runtime.h>
#include <cstddef>

// ---------------------------------------------------------------------------
// Explicit-duration HSMM forward DP on gfx1250 (MI455X).
//
// 64 workgroups (one batch element per WGP), 512 threads (16 wave32),
// sequential scan over T=2048 inside the kernel. Per step t (5 barriers):
//
//   Phase A : m = max4(wmax[]);  ph[i] = f16( exp(alpha[i] - m) )   (thr 0..127)
//   Phase B : trans_lin = p (1x128) * expA (128x128) via WMMA       (waves 0..7)
//             expA: constant f16 B-fragments in VGPRs; A fragment row 0 only,
//             2x ds_load_b128 per K-tile, 4 accumulating WMMAs.
//   Phase C : entry = log(trans_lin)+m (pi at t=0); g[t]=entry-cum[t];
//             push g into MIRRORED ring (rows r, r+128); cum += log_B[t].
//   Phase D : per (state j, duration chunk q): w[k]=fma(g,log2e,durlp2[k]),
//             local max mloc + local sum  s = sum exp2(w - mloc)
//             (32 immediate-offset DS loads, register-resident two passes).
//   Phase D3: combine 4 (mloc,s) pairs: M=max, S=sum s_q*exp2(m_q-M);
//             alpha[j] = fma(M + log2(S), ln2, cum[t+1]); wave-shuffle max
//             of alpha -> wmax[wid] (waves 0..3), consumed next step.
//
// All log/exp in the exp2/log2 domain via raw v_exp_f32 / v_log_f32
// (inputs guaranteed normal), folding the ln2 conversion into FMAs.
// d > t+1 masking: ring init to NEG=-1e30 -> exp2 underflows to 0 exactly.
// HBM traffic ~67MB total (log_B once) -> ~3us at 23.3 TB/s; the kernel is
// LDS/VALU bound (~2.1G logsumexp terms), so all DP state lives in LDS.
// ---------------------------------------------------------------------------

typedef __attribute__((ext_vector_type(16))) _Float16 v16h;
typedef __attribute__((ext_vector_type(8)))  _Float16 v8h;
typedef __attribute__((ext_vector_type(8)))  float    v8f;

#define BN   64
#define TN   2048
#define SN   128
#define DM   128
#define NEGV (-1.0e30f)
#define L2E  1.4426950408889634f   // log2(e)
#define LN2  0.6931471805599453f

__launch_bounds__(512, 1)
__global__ void hsmm_fwd_kernel(const float* __restrict__ logB,   // (B,T,S)
                                const float* __restrict__ pi,     // (S)
                                const float* __restrict__ A,      // (S,S)
                                const float* __restrict__ Dlg,    // (S,DM)
                                float* __restrict__ out)          // (B)
{
    // Mirrored ring: g[t'] at rows (t'&127) and (t'&127)+128, so the read
    // window [(t&127)+1 .. (t&127)+128] is contiguous (immediate DS offsets).
    __shared__ alignas(16) float    gring[2 * DM * SN];   // 128 KB
    __shared__ alignas(16) float    cumrun[SN];
    __shared__ alignas(16) float    alpha_s[SN];
    __shared__ alignas(16) float    translin[SN];
    __shared__ alignas(16) _Float16 ph[SN];               // p[] as f16
    __shared__ alignas(16) float    pmax[4 * SN];
    __shared__ alignas(16) float    psum[4 * SN];
    __shared__ float wmax[4];                             // per-wave alpha max

    const int tid  = threadIdx.x;
    const int lane = tid & 31;
    const int wid  = tid >> 5;
    const int b    = blockIdx.x;
    const int j    = tid & 127;   // state
    const int q    = tid >> 7;    // duration chunk 0..3 (32 durations each)

    // ---- init ----
    for (int i = tid; i < 2 * DM * SN; i += 512) gring[i] = NEGV;
    if (tid < SN) cumrun[tid] = 0.0f;

    // Duration log-probs, pre-scaled to the exp2 domain, in 32 VGPRs.
    float durlp2[32];
#pragma unroll
    for (int k = 0; k < 32; ++k) durlp2[k] = Dlg[j * DM + q * 32 + k] * L2E;

    // Constant expA B-operand WMMA fragments (f16), waves 0..7: N-tile = wid.
    //   lanes 0..15 : col = lane,    half h -> K = kt*32 + h
    //   lanes 16..31: col = lane-16, half h -> K = kt*32 + 16 + h
    v16h bf0, bf1, bf2, bf3;
    if (wid < 8) {
        const int col  = wid * 16 + (lane & 15);
        const int koff = (lane < 16) ? 0 : 16;
#pragma unroll
        for (int h = 0; h < 16; ++h) {
            bf0[h] = (_Float16)__expf(A[(0 * 32 + koff + h) * SN + col]);
            bf1[h] = (_Float16)__expf(A[(1 * 32 + koff + h) * SN + col]);
            bf2[h] = (_Float16)__expf(A[(2 * 32 + koff + h) * SN + col]);
            bf3[h] = (_Float16)__expf(A[(3 * 32 + koff + h) * SN + col]);
        }
    }
    __syncthreads();

    float m_al = 0.0f;   // max_j alpha (threads 0..127, valid from t=1 on)

    for (int t = 0; t < TN; ++t) {
        if (t > 0) {
            // -------- Phase A: p = exp(alpha - m) --------
            if (tid < SN) {
                m_al = fmaxf(fmaxf(wmax[0], wmax[1]), fmaxf(wmax[2], wmax[3]));
                ph[tid] = (_Float16)__builtin_amdgcn_exp2f(
                    (alpha_s[tid] - m_al) * L2E);
            }
            __syncthreads();

            // -------- Phase B: trans_lin = p * expA via WMMA --------
            if (wid < 8) {                        // wave-uniform: EXEC all 1s
                // A fragment row 0 only (lanes 0 / 16); other rows are
                // don't-care so every lane just loads its half-group chunk.
                const v8h* php = (const v8h*)ph;  // 16-byte chunks of 8 halves
                const int  sel = (lane & 16) ? 2 : 0;
                const v8h lo0 = php[0  + sel], hi0 = php[1  + sel];
                const v8h lo1 = php[4  + sel], hi1 = php[5  + sel];
                const v8h lo2 = php[8  + sel], hi2 = php[9  + sel];
                const v8h lo3 = php[12 + sel], hi3 = php[13 + sel];
                v8f acc = {0.f, 0.f, 0.f, 0.f, 0.f, 0.f, 0.f, 0.f};
#define MKAF(lo, hi) __builtin_shufflevector(lo, hi, 0,1,2,3,4,5,6,7, \
                                             8,9,10,11,12,13,14,15)
                acc = __builtin_amdgcn_wmma_f32_16x16x32_f16(
                    false, MKAF(lo0, hi0), false, bf0, (short)0, acc, false, false);
                acc = __builtin_amdgcn_wmma_f32_16x16x32_f16(
                    false, MKAF(lo1, hi1), false, bf1, (short)0, acc, false, false);
                acc = __builtin_amdgcn_wmma_f32_16x16x32_f16(
                    false, MKAF(lo2, hi2), false, bf2, (short)0, acc, false, false);
                acc = __builtin_amdgcn_wmma_f32_16x16x32_f16(
                    false, MKAF(lo3, hi3), false, bf3, (short)0, acc, false, false);
#undef MKAF
                // D row 0 lives in acc[0] on lanes 0..15 (N = lane).
                if (lane < 16) translin[wid * 16 + lane] = acc[0];
            }
            __syncthreads();
        }

        // -------- Phase C: entry, mirrored g-ring push, cumsum update -----
        if (tid < SN) {
            const size_t base = ((size_t)b * TN + t) * SN;
            const float  lb   = logB[base + tid];
            float entry;
            if (t == 0) {
                entry = pi[tid];
            } else {
                // branch-free guard: underflowed transition ~ e^-69 weight.
                // raw v_log_f32 (log2) on a guaranteed-normal input; ln2
                // conversion folded into the FMA with m.
                entry = __builtin_fmaf(
                    __builtin_amdgcn_logf(fmaxf(translin[tid], 1.0e-30f)),
                    LN2, m_al);
            }
            const float g = entry - cumrun[tid];     // g[t] = entry[t]-cum[t]
            const int   r = t & 127;
            gring[r * SN + tid]         = g;
            gring[(r + 128) * SN + tid] = g;         // mirror copy
            cumrun[tid] += lb;                       // now cum[t+1]
            if (t + 1 < TN)                          // global_prefetch_b8
                __builtin_prefetch(logB + base + SN + tid, 0, 1);
        }
        __syncthreads();

        // -------- Phase D: chunk-local max + exp2-sum (one barrier) -------
        // Duration d = q*32 + k + 1 -> row = (t&127) + 128 - q*32 - k.
        const int baseidx = ((t & 127) + 128 - q * 32 - 31) * SN + j;
        float w32[32];
        float mloc = NEGV;
#pragma unroll
        for (int k = 0; k < 32; ++k) {
            const float g = gring[baseidx + (31 - k) * SN];
            w32[k] = __builtin_fmaf(g, L2E, durlp2[k]);   // exp2 domain
            mloc   = fmaxf(mloc, w32[k]);
        }
        float s = 0.0f;
#pragma unroll
        for (int k = 0; k < 32; ++k)
            s += __builtin_amdgcn_exp2f(w32[k] - mloc);   // exp2(-huge)==0
        pmax[q * SN + j] = mloc;
        psum[q * SN + j] = s;
        __syncthreads();

        // -------- Phase D3: combine chunks, alpha[j], per-wave max --------
        if (tid < SN) {
            const float m0 = pmax[0 * SN + tid], m1 = pmax[1 * SN + tid];
            const float m2 = pmax[2 * SN + tid], m3 = pmax[3 * SN + tid];
            const float Mw = fmaxf(fmaxf(m0, m1), fmaxf(m2, m3));
            float an = NEGV;
            if (Mw > -1.0e29f) {
                const float S4 =
                    psum[0 * SN + tid] * __builtin_amdgcn_exp2f(m0 - Mw) +
                    psum[1 * SN + tid] * __builtin_amdgcn_exp2f(m1 - Mw) +
                    psum[2 * SN + tid] * __builtin_amdgcn_exp2f(m2 - Mw) +
                    psum[3 * SN + tid] * __builtin_amdgcn_exp2f(m3 - Mw);
                // S4 >= 1 (max chunk contributes >= 1): raw log2 is safe.
                an = __builtin_fmaf(Mw + __builtin_amdgcn_logf(S4),
                                    LN2, cumrun[tid]);
            }
            alpha_s[tid] = an;
            // per-wave (waves 0..3) shuffle max of alpha
            float wm = an;
            wm = fmaxf(wm, __shfl_xor(wm, 16));
            wm = fmaxf(wm, __shfl_xor(wm, 8));
            wm = fmaxf(wm, __shfl_xor(wm, 4));
            wm = fmaxf(wm, __shfl_xor(wm, 2));
            wm = fmaxf(wm, __shfl_xor(wm, 1));
            if (lane == 0) wmax[wid] = wm;
        }
        __syncthreads();
    }

    // -------- final: loglik[b] = logsumexp_j alpha[T-1, j] --------
    if (tid < SN) {
        const float m = fmaxf(fmaxf(wmax[0], wmax[1]), fmaxf(wmax[2], wmax[3]));
        psum[tid] = __builtin_amdgcn_exp2f((alpha_s[tid] - m) * L2E);
        pmax[0]   = m;   // all writers store the same value
    }
    __syncthreads();
    if (tid == 0) {
        float acc = 0.0f;
        for (int i = 0; i < SN; ++i) acc += psum[i];
        out[b] = __builtin_fmaf(__builtin_amdgcn_logf(acc), LN2, pmax[0]);
    }
}

extern "C" void kernel_launch(void* const* d_in, const int* in_sizes, int n_in,
                              void* d_out, int out_size, void* d_ws, size_t ws_size,
                              hipStream_t stream) {
    (void)in_sizes; (void)n_in; (void)ws_size; (void)d_ws; (void)out_size;
    const float* logB = (const float*)d_in[0];   // (64,2048,128) f32
    const float* pi   = (const float*)d_in[1];   // (128)         f32
    const float* A    = (const float*)d_in[2];   // (128,128)     f32
    const float* Dlg  = (const float*)d_in[3];   // (128,128)     f32
    float*       out  = (float*)d_out;           // (64)          f32
    hipLaunchKernelGGL(hsmm_fwd_kernel, dim3(BN), dim3(512), 0, stream,
                       logB, pi, A, Dlg, out);
}
